// Classifier_78185584656978
// MI455X (gfx1250) — matrix operand order repaired and checked by reading, not compiled, not run on hardware
//
#include <hip/hip_runtime.h>
#include <hip/hip_bf16.h>

#define NN      200000
#define NE      3200000
#define NGRAPH  100
#define NPG     2000
#define D       128
#define DOUT    64

typedef __attribute__((ext_vector_type(16))) __bf16        v16bf;
typedef __attribute__((ext_vector_type(8)))  float         v8f;
typedef __attribute__((ext_vector_type(8)))  unsigned int  v8u;

union BfPack {
    unsigned short us[16];
    v16bf          bf;
    v8u            u8;
};

__device__ __forceinline__ unsigned short f32_to_bf16(float f) {
    unsigned int u = __float_as_uint(f);
    u += 0x7FFFu + ((u >> 16) & 1u);        // round-to-nearest-even
    return (unsigned short)(u >> 16);
}

// ---------------- degree + norm ----------------
__global__ void degree_kernel(const int* __restrict__ dst, float* __restrict__ deg, int ne) {
    int t = blockIdx.x * blockDim.x + threadIdx.x;
    if (t < ne) atomicAdd(&deg[dst[t]], 1.0f);
}

__global__ void norm_kernel(float* __restrict__ deg, int n) {
    int t = blockIdx.x * blockDim.x + threadIdx.x;
    if (t < n) deg[t] = rsqrtf(fmaxf(deg[t], 1.0f));
}

// ---------------- weight prepack: f32 [K=128][N=128] -> bf16 WMMA B-fragment order ----
// packed[ (nt*4+ks)*32 + lane ][ v ]  = w[ ks*32 + (v>>3)*16 + (lane>>4)*8 + (v&7) ][ nt*16 + (lane&15) ]
__global__ void prepack_kernel(const float* __restrict__ w, unsigned short* __restrict__ out) {
    int idx = blockIdx.x * blockDim.x + threadIdx.x;          // 0 .. 16383
    if (idx >= 8 * 4 * 32 * 16) return;
    int v    = idx & 15;
    int lane = (idx >> 4) & 31;
    int ks   = (idx >> 9) & 3;
    int nt   = (idx >> 11) & 7;
    int n = nt * 16 + (lane & 15);
    int k = ks * 32 + (v >> 3) * 16 + (lane >> 4) * 8 + (v & 7);
    out[idx] = f32_to_bf16(w[k * D + n]);
}

// ---------------- edge scatter: agg[dst] += h[src] * norm[src] -------------------------
// one wave32 per edge, 4 dims per lane (512B contiguous row per edge)
__global__ void scatter_kernel(const float* __restrict__ h, const float* __restrict__ nrm,
                               const int* __restrict__ src, const int* __restrict__ dst,
                               float* __restrict__ agg, int ne) {
    long long t = (long long)blockIdx.x * blockDim.x + threadIdx.x;
    int e = (int)(t >> 5);
    if (e >= ne) return;
    int lane = (int)(t & 31);
    int s = src[e], d = dst[e];
    float nv = nrm[s];
    float4 hv = *(const float4*)(h + (size_t)s * D + lane * 4);
    float* ap = agg + (size_t)d * D + lane * 4;
    atomicAdd(ap + 0, hv.x * nv);
    atomicAdd(ap + 1, hv.y * nv);
    atomicAdd(ap + 2, hv.z * nv);
    atomicAdd(ap + 3, hv.w * nv);
}

// ---------------- node GEMM via WMMA bf16: out = relu((agg @ W)*norm + b) --------------
// 8 waves/block, each wave computes a 16-row x 128-col tile: 8 ntiles x 4 ksteps WMMA
__global__ void gemm_kernel(const float* __restrict__ agg, const unsigned short* __restrict__ wpk,
                            const float* __restrict__ bias, const float* __restrict__ nrm,
                            float* __restrict__ out, int nrows) {
    int lane = threadIdx.x & 31;
    int wv   = threadIdx.x >> 5;
    int row0 = blockIdx.x * 128 + wv * 16;
    if (row0 >= nrows) return;                    // wave-uniform: EXEC stays all-ones
    int half = lane >> 4;                         // 0 | 1
    int l16  = lane & 15;

    // ---- A fragments: 16x32 bf16 per kstep, ISA lane layout ----
    v16bf afrag[4];
    const float4* arow = (const float4*)(agg + (size_t)(row0 + l16) * D);
    #pragma unroll
    for (int ks = 0; ks < 4; ++ks) {
        int kb = ks * 32 + half * 8;              // floats
        float4 p0 = arow[(kb >> 2) + 0];
        float4 p1 = arow[(kb >> 2) + 1];
        float4 q0 = arow[(kb >> 2) + 4];          // +16 floats
        float4 q1 = arow[(kb >> 2) + 5];
        BfPack pa;
        pa.us[0]  = f32_to_bf16(p0.x); pa.us[1]  = f32_to_bf16(p0.y);
        pa.us[2]  = f32_to_bf16(p0.z); pa.us[3]  = f32_to_bf16(p0.w);
        pa.us[4]  = f32_to_bf16(p1.x); pa.us[5]  = f32_to_bf16(p1.y);
        pa.us[6]  = f32_to_bf16(p1.z); pa.us[7]  = f32_to_bf16(p1.w);
        pa.us[8]  = f32_to_bf16(q0.x); pa.us[9]  = f32_to_bf16(q0.y);
        pa.us[10] = f32_to_bf16(q0.z); pa.us[11] = f32_to_bf16(q0.w);
        pa.us[12] = f32_to_bf16(q1.x); pa.us[13] = f32_to_bf16(q1.y);
        pa.us[14] = f32_to_bf16(q1.z); pa.us[15] = f32_to_bf16(q1.w);
        afrag[ks] = pa.bf;
    }

    // per-lane row norms for epilogue (C layout: lane half h holds rows h*8 .. h*8+7)
    float rn[8];
    #pragma unroll
    for (int r = 0; r < 8; ++r) rn[r] = nrm[row0 + half * 8 + r];

    #pragma unroll
    for (int nt = 0; nt < 8; ++nt) {
        int n0 = nt * 16;
        v8f acc = {};
        #pragma unroll
        for (int ks = 0; ks < 4; ++ks) {
            BfPack pb;
            pb.u8 = *(const v8u*)(wpk + ((nt * 4 + ks) * 32 + lane) * 16);
            acc = __builtin_amdgcn_wmma_f32_16x16x32_bf16(
                false, afrag[ks], false, pb.bf, (short)0, acc, false, false);
        }
        float bcol = bias[n0 + l16];
        #pragma unroll
        for (int r = 0; r < 8; ++r) {
            int m = half * 8 + r;
            float v = acc[r] * rn[r] + bcol;
            out[(size_t)(row0 + m) * D + n0 + l16] = fmaxf(v, 0.0f);
        }
    }
}

// ---------------- readout: out[g] = h[to_fetch[g] + g*NPG] @ w3^T + b3 -----------------
__global__ void final_kernel(const float* __restrict__ h, const int* __restrict__ tf,
                             const float* __restrict__ w3, const float* __restrict__ b3,
                             float* __restrict__ out) {
    int g = blockIdx.x;
    int o = threadIdx.x;                          // 64 threads
    int node = tf[g] + g * NPG;
    const float4* hr = (const float4*)(h + (size_t)node * D);
    const float4* wr = (const float4*)(w3 + o * D);
    float s = 0.0f;
    #pragma unroll 8
    for (int k = 0; k < D / 4; ++k) {
        float4 a = hr[k], b = wr[k];
        s += a.x * b.x + a.y * b.y + a.z * b.z + a.w * b.w;
    }
    out[g * DOUT + o] = s + b3[o];
}

extern "C" void kernel_launch(void* const* d_in, const int* in_sizes, int n_in,
                              void* d_out, int out_size, void* d_ws, size_t ws_size,
                              hipStream_t stream) {
    const float* features = (const float*)d_in[0];
    const int*   src      = (const int*)d_in[1];
    const int*   dst      = (const int*)d_in[2];
    const int*   to_fetch = (const int*)d_in[3];
    const float* w1       = (const float*)d_in[4];
    const float* b1       = (const float*)d_in[5];
    const float* w2       = (const float*)d_in[6];
    const float* b2       = (const float*)d_in[7];
    const float* w3       = (const float*)d_in[8];
    const float* b3       = (const float*)d_in[9];
    float*       outp     = (float*)d_out;

    char* ws = (char*)d_ws;
    float*          nrm  = (float*)ws;                               // 800 KB
    unsigned short* wpk1 = (unsigned short*)(ws + (1ull << 20));     // 32 KB
    unsigned short* wpk2 = wpk1 + 16384;                             // 32 KB
    float*          bufA = (float*)(ws + (2ull << 20));              // 102.4 MB (agg)
    float*          bufB = (float*)(ws + (2ull << 20) + 104857600ull); // 102.4 MB (h)

    const size_t featBytes = (size_t)NN * D * sizeof(float);
    const int scatterBlocks = (int)(((long long)NE * 32 + 255) / 256);
    const int gemmBlocks    = (NN + 127) / 128;

    // degrees -> norm
    hipMemsetAsync(nrm, 0, (size_t)NN * sizeof(float), stream);
    degree_kernel<<<(NE + 255) / 256, 256, 0, stream>>>(dst, nrm, NE);
    norm_kernel<<<(NN + 255) / 256, 256, 0, stream>>>(nrm, NN);

    // bf16 fragment-order weight packs
    prepack_kernel<<<64, 256, 0, stream>>>(w1, wpk1);
    prepack_kernel<<<64, 256, 0, stream>>>(w2, wpk2);

    // layer 1
    hipMemsetAsync(bufA, 0, featBytes, stream);
    scatter_kernel<<<scatterBlocks, 256, 0, stream>>>(features, nrm, src, dst, bufA, NE);
    gemm_kernel<<<gemmBlocks, 256, 0, stream>>>(bufA, wpk1, b1, nrm, bufB, NN);

    // layer 2
    hipMemsetAsync(bufA, 0, featBytes, stream);
    scatter_kernel<<<scatterBlocks, 256, 0, stream>>>(bufB, nrm, src, dst, bufA, NE);
    gemm_kernel<<<gemmBlocks, 256, 0, stream>>>(bufA, wpk2, b2, nrm, bufB, NN);

    // readout
    final_kernel<<<NGRAPH, DOUT, 0, stream>>>(bufB, to_fetch, w3, b3, outp);
}
